// MFM_76897094468447
// MI455X (gfx1250) — matrix operand order repaired and checked by reading, not compiled, not run on hardware
//
#include <hip/hip_runtime.h>
#include <math.h>

#define NTOK 65536   // B*H*W = 16*64*64
#define HDIM 64
#define BDIM 16

typedef __attribute__((ext_vector_type(16))) __bf16 bf16x16;
typedef __attribute__((ext_vector_type(8)))  float  f32x8;

// ---------------------------------------------------------------- activations
__device__ __forceinline__ float actf(float v, int act, int col, int O) {
  switch (act) {
    case 1: v = fmaxf(v, 0.f); break;                                   // relu
    case 2: v = v / (1.f + __expf(-v)); break;                          // silu
    case 3: v = 0.5f * v * (1.f + erff(v * 0.70710678118f)); break;     // gelu (exact)
    case 4: v = 1.f / (1.f + __expf(-v)); break;                        // sigmoid
    case 5: v = (v > 0.f) ? v + 1.f : __expf(v); break;                 // elu+1
    case 6: if (col < (O >> 1)) v = (v > 0.f) ? v + 1.f : __expf(v);    // elu+1 on k-half
            break;
    default: break;
  }
  return v;
}

// ---------------------------------------------------------------- WMMA frags
// A 16x32 bf16: lane<16 -> K=0..15, lane>=16 -> K=16..31 (16 contiguous).  p = &X[row*ld + k0 + koffA]
__device__ __forceinline__ bf16x16 frag_a(const float* __restrict__ p) {
  bf16x16 r;
#pragma unroll
  for (int i = 0; i < 8; ++i) r[i] = (__bf16)p[i];
#pragma unroll
  for (int i = 0; i < 8; ++i) r[i + 8] = (__bf16)p[8 + i];
  return r;
}
// B 32x16 bf16: lane<16 -> K=0..15, lane>=16 -> K=16..31 (16 contiguous).
__device__ __forceinline__ bf16x16 frag_b(const float* __restrict__ p) {
  bf16x16 r;
#pragma unroll
  for (int i = 0; i < 16; ++i) r[i] = (__bf16)p[i];
  return r;
}

// Y[M,O] = act(((X[M,K] @ W'[O,:]^T + pre) + b1)*scale*scaleK + b2) + post1 + post2
// W row stride = ldw, column window starts at wOff (for split-K concat GEMMs).
// Requires M%16==0, K%32==0, O%64==0.
__global__ void k_gemm_wmma(const float* __restrict__ X, const float* __restrict__ W,
                            const float* __restrict__ pre, const float* __restrict__ b1,
                            const float* __restrict__ scale, float scaleK,
                            const float* __restrict__ b2,
                            const float* __restrict__ post1, const float* __restrict__ post2,
                            float* __restrict__ Y, int M, int K, int O, int ldw, int wOff,
                            int act) {
  int wid  = (blockIdx.x * blockDim.x + threadIdx.x) >> 5;
  int lane = threadIdx.x & 31;
  int tn4cnt = O >> 6;
  int tm  = wid / tn4cnt;
  int tn4 = wid - tm * tn4cnt;
  if (tm * 16 >= M) return;

  int rowA    = tm * 16 + (lane & 15);
  int koffA   = (lane < 16) ? 0 : 16;
  int koffB   = (lane < 16) ? 0 : 16;
  int colLane = lane & 15;
  const float* aRow  = X + (size_t)rowA * K;
  const float* bBase = W + (size_t)wOff;

  f32x8 zero = {0.f, 0.f, 0.f, 0.f, 0.f, 0.f, 0.f, 0.f};
  f32x8 acc[4];
#pragma unroll
  for (int j = 0; j < 4; ++j) acc[j] = zero;

  for (int k0 = 0; k0 < K; k0 += 32) {
    if (k0 + 32 < K) __builtin_prefetch(aRow + k0 + 32, 0, 0);
    bf16x16 a = frag_a(aRow + k0 + koffA);
#pragma unroll
    for (int j = 0; j < 4; ++j) {
      int colW = tn4 * 64 + j * 16 + colLane;
      bf16x16 bf = frag_b(bBase + (size_t)colW * ldw + k0 + koffB);
      acc[j] = __builtin_amdgcn_wmma_f32_16x16x32_bf16(false, a, false, bf,
                                                       (short)0, acc[j], false, false);
    }
  }

  int rowD = tm * 16 + ((lane < 16) ? 0 : 8);
#pragma unroll
  for (int j = 0; j < 4; ++j) {
    int col = tn4 * 64 + j * 16 + colLane;
    float sc  = scale ? scale[col] * scaleK : 1.f;
    float bb1 = b1 ? b1[col] : 0.f;
    float bb2 = b2 ? b2[col] : 0.f;
#pragma unroll
    for (int r = 0; r < 8; ++r) {
      size_t idx = (size_t)(rowD + r) * O + col;
      float v = acc[j][r];
      if (pre) v += pre[idx];
      v = (v + bb1) * sc + bb2;
      v = actf(v, act, col, O);
      if (post1) v += post1[idx];
      if (post2) v += post2[idx];
      Y[idx] = v;
    }
  }
}

// Small/irregular GEMM (gate branch: O=8 or K=8).
__global__ void k_gemm_naive(const float* __restrict__ X, const float* __restrict__ W,
                             const float* __restrict__ b1, const float* __restrict__ scale,
                             float scaleK, const float* __restrict__ b2,
                             float* __restrict__ Y, int M, int K, int O, int act) {
  int id = blockIdx.x * blockDim.x + threadIdx.x;
  if (id >= M * O) return;
  int o = id % O, t = id / O;
  const float* row = X + (size_t)t * K;
  const float* w   = W + (size_t)o * K;
  float v = 0.f;
  for (int k = 0; k < K; ++k) v += row[k] * w[k];
  float sc = scale ? scale[o] * scaleK : 1.f;
  v = (v + (b1 ? b1[o] : 0.f)) * sc + (b2 ? b2[o] : 0.f);
  Y[(size_t)t * O + o] = actf(v, act, o, O);
}

// ---------------------------------------------------------------- depthwise 3x3
__global__ void k_dwconv3(const float* __restrict__ X, const float* __restrict__ Wt,
                          const float* __restrict__ bias, float* __restrict__ Y,
                          int C, int addInput, int act) {
  int id = blockIdx.x * blockDim.x + threadIdx.x;
  if (id >= NTOK * C) return;
  int c = id % C, t = id / C;
  int xx = t & 63, yy = (t >> 6) & 63, b = t >> 12;
  const float* wk = Wt + c * 9;
  float acc = bias ? bias[c] : 0.f;
#pragma unroll
  for (int ky = 0; ky < 3; ++ky) {
    int ys = yy + ky - 1;
    if ((unsigned)ys >= 64u) continue;
#pragma unroll
    for (int kx = 0; kx < 3; ++kx) {
      int xs = xx + kx - 1;
      if ((unsigned)xs >= 64u) continue;
      acc += X[((size_t)((b * 64 + ys) * 64 + xs)) * C + c] * wk[ky * 3 + kx];
    }
  }
  if (addInput) acc += X[(size_t)t * C + c];
  if (act == 2) acc = acc / (1.f + __expf(-acc));
  Y[(size_t)t * C + c] = acc;
}

// grouped conv: 128 in, 64 out, groups=64 (2 in-ch per out-ch), no bias. ks=3 or 5.
__global__ void k_groupconv(const float* __restrict__ X, const float* __restrict__ Wt,
                            float* __restrict__ Y, int ks) {
  int id = blockIdx.x * blockDim.x + threadIdx.x;
  if (id >= NTOK * 64) return;
  int o = id % 64, t = id / 64;
  int xx = t & 63, yy = (t >> 6) & 63, b = t >> 12;
  int pad = ks >> 1;
  float acc = 0.f;
  for (int i = 0; i < 2; ++i) {
    int ci = 2 * o + i;
    const float* wk = Wt + ((o * 2 + i) * ks) * ks;
    for (int ky = 0; ky < ks; ++ky) {
      int ys = yy + ky - pad;
      if ((unsigned)ys >= 64u) continue;
      for (int kx = 0; kx < ks; ++kx) {
        int xs = xx + kx - pad;
        if ((unsigned)xs >= 64u) continue;
        acc += X[((size_t)((b * 64 + ys) * 64 + xs)) * 128 + ci] * wk[ky * ks + kx];
      }
    }
  }
  Y[(size_t)t * 64 + o] = acc;
}

// ---------------------------------------------------------------- layernorm (C=128), wave32/token
__global__ void k_layernorm(const float* __restrict__ X, const float* __restrict__ g,
                            const float* __restrict__ b, float* __restrict__ Y, int M) {
  int wave = (blockIdx.x * blockDim.x + threadIdx.x) >> 5;
  int lane = threadIdx.x & 31;
  if (wave >= M) return;
  const float* row = X + (size_t)wave * 128;
  float v[4], s = 0.f, s2 = 0.f;
#pragma unroll
  for (int i = 0; i < 4; ++i) { v[i] = row[lane + 32 * i]; s += v[i]; s2 += v[i] * v[i]; }
#pragma unroll
  for (int m = 16; m >= 1; m >>= 1) { s += __shfl_xor(s, m, 32); s2 += __shfl_xor(s2, m, 32); }
  float mean = s * (1.f / 128.f);
  float var  = s2 * (1.f / 128.f) - mean * mean;
  float inv  = rsqrtf(var + 1e-5f);
#pragma unroll
  for (int i = 0; i < 4; ++i) {
    int c = lane + 32 * i;
    Y[(size_t)wave * 128 + c] = (v[i] - mean) * inv * g[c] + b[c];
  }
}

// ---------------------------------------------------------------- 2-D DFT (64x64) per (b,c) plane
__global__ void k_fft2d_fwd(const float* __restrict__ X, float* __restrict__ fRe,
                            float* __restrict__ fIm, int C) {
  __shared__ float sIn[4096], sRe[4160], sIm[4160], twc[64], tws[64];
  int plane = blockIdx.x, b = plane / C, c = plane % C, tid = threadIdx.x;
  if (tid < 64) {
    float ang = -6.283185307f * (float)tid / 64.f;
    twc[tid] = cosf(ang); tws[tid] = sinf(ang);
  }
  size_t pbase = (size_t)b * 4096 * C + c;
  for (int n = tid; n < 4096; n += 256) sIn[n] = X[pbase + (size_t)n * C];
  __syncthreads();
  for (int n = tid; n < 4096; n += 256) {               // row DFT (real input)
    int r = n >> 6, j = n & 63;
    const float* rp = &sIn[r * 64];
    float re = 0.f, im = 0.f;
    for (int k = 0; k < 64; ++k) {
      int m = (j * k) & 63;
      float x = rp[k];
      re += x * twc[m]; im += x * tws[m];
    }
    sRe[r * 65 + j] = re; sIm[r * 65 + j] = im;
  }
  __syncthreads();
  for (int n = tid; n < 4096; n += 256) {               // col DFT
    int i = n >> 6, j = n & 63;
    float re = 0.f, im = 0.f;
    for (int r = 0; r < 64; ++r) {
      int m = (i * r) & 63;
      float ar = sRe[r * 65 + j], ai = sIm[r * 65 + j];
      float cw = twc[m], sw = tws[m];
      re += ar * cw - ai * sw; im += ar * sw + ai * cw;
    }
    fRe[pbase + (size_t)n * C] = re;
    fIm[pbase + (size_t)n * C] = im;
  }
}

// inverse DFT of (gate * f), fused |.| -> BN -> ReLU epilogue
__global__ void k_ifft2d_post(const float* __restrict__ fRe, const float* __restrict__ fIm,
                              const float* __restrict__ gate, const float* __restrict__ nbg,
                              const float* __restrict__ nbb, float* __restrict__ out, int C) {
  __shared__ float sRe[4160], sIm[4160], twc[64], tws[64];
  int plane = blockIdx.x, b = plane / C, c = plane % C, tid = threadIdx.x;
  if (tid < 64) {
    float ang = 6.283185307f * (float)tid / 64.f;       // conjugate twiddles
    twc[tid] = cosf(ang); tws[tid] = sinf(ang);
  }
  __syncthreads();
  size_t pbase = (size_t)b * 4096 * C + c;
  for (int n = tid; n < 4096; n += 256) {               // row pass, gate fused on load
    int r = n >> 6, j = n & 63;
    size_t rb = pbase + (size_t)(r * 64) * C;
    float re = 0.f, im = 0.f;
    for (int k = 0; k < 64; ++k) {
      int m = (j * k) & 63;
      size_t ix = rb + (size_t)k * C;
      float g  = gate[ix];
      float ar = g * fRe[ix], ai = g * fIm[ix];
      float cw = twc[m], sw = tws[m];
      re += ar * cw - ai * sw; im += ar * sw + ai * cw;
    }
    sRe[r * 65 + j] = re; sIm[r * 65 + j] = im;
  }
  __syncthreads();
  float s1 = nbg[c] * rsqrtf(1.f + 1e-5f);
  float b1 = nbb[c];
  for (int n = tid; n < 4096; n += 256) {               // col pass + epilogue
    int i = n >> 6, j = n & 63;
    float re = 0.f, im = 0.f;
    for (int r = 0; r < 64; ++r) {
      int m = (i * r) & 63;
      float ar = sRe[r * 65 + j], ai = sIm[r * 65 + j];
      float cw = twc[m], sw = tws[m];
      re += ar * cw - ai * sw; im += ar * sw + ai * cw;
    }
    re *= (1.f / 4096.f); im *= (1.f / 4096.f);
    float y = sqrtf(re * re + im * im) * s1 + b1;
    out[pbase + (size_t)n * C] = fmaxf(y, 0.f);
  }
}

// ---------------------------------------------------------------- linear attention
// kv: [N,128] (k=cols 0..63 already elu+1, v=cols 64..127). One block per (b,h).
__global__ void k_attn_reduce(const float* __restrict__ kv, float* __restrict__ ksum,
                              float* __restrict__ ctx, int Nn) {
  int b = blockIdx.x >> 3, h = blockIdx.x & 7;
  const float* base = kv + (size_t)b * Nn * 128;
  float lk[8] = {0}, lc[8][8] = {{0}};
  for (int n = threadIdx.x; n < Nn; n += blockDim.x) {
    const float* row = base + (size_t)n * 128;
    float kk[8], vv[8];
#pragma unroll
    for (int d = 0; d < 8; ++d) { kk[d] = row[h * 8 + d]; vv[d] = row[64 + h * 8 + d]; }
#pragma unroll
    for (int d = 0; d < 8; ++d) {
      lk[d] += kk[d];
#pragma unroll
      for (int e = 0; e < 8; ++e) lc[d][e] += kk[d] * vv[e];
    }
  }
  __shared__ float red[72];
  if (threadIdx.x < 72) red[threadIdx.x] = 0.f;
  __syncthreads();
#pragma unroll
  for (int d = 0; d < 8; ++d) atomicAdd(&red[d], lk[d]);
#pragma unroll
  for (int d = 0; d < 8; ++d)
#pragma unroll
    for (int e = 0; e < 8; ++e) atomicAdd(&red[8 + d * 8 + e], lc[d][e]);
  __syncthreads();
  if (threadIdx.x < 8)  ksum[(b * 8 + h) * 8 + threadIdx.x] = red[threadIdx.x];
  if (threadIdx.x < 64) ctx[(b * 8 + h) * 64 + threadIdx.x] = red[8 + threadIdx.x];
}

__global__ void k_attn_apply(const float* __restrict__ q, const float* __restrict__ ksum,
                             const float* __restrict__ ctx, float* __restrict__ out, int Nn) {
  int id = blockIdx.x * blockDim.x + threadIdx.x;
  if (id >= BDIM * Nn * 8) return;
  int h = id & 7, t = id >> 3, b = t / Nn;
  const float* qr = q + (size_t)t * 64 + h * 8;
  const float* ks = ksum + (b * 8 + h) * 8;
  const float* cx = ctx + (b * 8 + h) * 64;
  float qq[8], den = 0.f;
#pragma unroll
  for (int d = 0; d < 8; ++d) { qq[d] = qr[d]; den += qq[d] * ks[d]; }
  float dinv = 1.f / fmaxf(den, 1e-6f);
#pragma unroll
  for (int e = 0; e < 8; ++e) {
    float s = 0.f;
#pragma unroll
    for (int d = 0; d < 8; ++d) s += qq[d] * cx[d * 8 + e];
    out[(size_t)t * 64 + h * 8 + e] = s * dinv;
  }
}

// Y[t,c] = concat(a3,a5)[t,c] * ar[t,c]
__global__ void k_concat_mul(const float* __restrict__ a3, const float* __restrict__ a5,
                             const float* __restrict__ ar, float* __restrict__ Y) {
  int id = blockIdx.x * blockDim.x + threadIdx.x;
  if (id >= NTOK * 128) return;
  int c = id & 127, t = id >> 7;
  float v = (c < 64) ? a3[(size_t)t * 64 + c] : a5[(size_t)t * 64 + (c - 64)];
  Y[(size_t)t * 128 + c] = v * ar[(size_t)t * 128 + c];
}

// ================================================================ host
enum {
  IN_X = 0, IN_CONV1_W, IN_CONV1_B, IN_CONV1_BN_G, IN_CONV1_BN_B,
  IN_CPE1_W, IN_CPE1_B, IN_CPE2_W, IN_CPE2_B,
  IN_W1_W, IN_W1_B, IN_WBN_G, IN_WBN_B, IN_W2_W, IN_W2_B,
  IN_NBN_G, IN_NBN_B, IN_LN1_G, IN_LN1_B, IN_LN2_G, IN_LN2_B,
  IN_DW3_PW_W, IN_DW3_PW_B, IN_DW3_W, IN_DW5_PW_W, IN_DW5_PW_B, IN_DW5_W,
  IN_ACTPROJ_W, IN_ACTPROJ_B, IN_INPROJ2_W, IN_INPROJ2_B, IN_DWC2_W, IN_DWC2_B,
  IN_Q_W, IN_Q_B, IN_KV_W, IN_KV_B, IN_PR_W, IN_PR_B,
  IN_OUTPROJ_W, IN_OUTPROJ_B, IN_FFN_W1, IN_FFN_W2, IN_PROJOUT_W,
  IN_REDUCE_W, IN_REDUCE_B, IN_REDUCE_BN_G, IN_REDUCE_BN_B
};

extern "C" void kernel_launch(void* const* d_in, const int* in_sizes, int n_in,
                              void* d_out, int out_size, void* d_ws, size_t ws_size,
                              hipStream_t stream) {
  (void)in_sizes; (void)n_in; (void)out_size; (void)ws_size;
  const float* P[48];
  for (int i = 0; i < 48; ++i) P[i] = (const float*)d_in[i];
  const float* x = P[IN_X];
  float* ws = (float*)d_ws;
  float* outp = (float*)d_out;

  const size_t S  = (size_t)NTOK * 128;
  const size_t Sh = (size_t)NTOK * 64;
  float* x0  = ws;            float* x1  = ws + S;      float* fmt = ws + 2 * S;
  float* bA  = ws + 3 * S;    float* bB  = ws + 4 * S;  float* bC  = ws + 5 * S;
  float* fRe = ws + 6 * S;    float* fIm = ws + 7 * S;  float* gate = ws + 8 * S;
  float* hA  = ws + 9 * S;    float* hB  = hA + Sh;
  float* hC  = ws + 10 * S;   float* hD  = hC + Sh;
  float* h1b = ws + 11 * S;                               // 4*S for FFN hidden
  float* gh    = ws + 15 * S;                             // [N,8]
  float* ksum3 = gh + (size_t)NTOK * 8;
  float* ctx3  = ksum3 + 1024;
  float* ksum5 = ctx3 + 8192;
  float* ctx5  = ksum5 + 1024;

  const float BNK = 1.f / sqrtf(1.f + 1e-5f);
  const int NP = BDIM * 128;                              // FFT planes

  auto gemm = [&](const float* X, const float* W, const float* pre, const float* b1,
                  const float* scale, float scaleK, const float* b2,
                  const float* post1, const float* post2, float* Y,
                  int K, int O, int ldw, int wOff, int act) {
    int waves  = (NTOK / 16) * (O / 64);
    int blocks = (waves + 7) / 8;
    k_gemm_wmma<<<blocks, 256, 0, stream>>>(X, W, pre, b1, scale, scaleK, b2,
                                            post1, post2, Y, NTOK, K, O, ldw, wOff, act);
  };
  auto fmt_pipe = [&](const float* src, float* dst) {
    k_fft2d_fwd<<<NP, 256, 0, stream>>>(src, fRe, fIm, 128);
    k_gemm_naive<<<(NTOK * 8 + 255) / 256, 256, 0, stream>>>(
        fRe, P[IN_W1_W], P[IN_W1_B], P[IN_WBN_G], BNK, P[IN_WBN_B], gh, NTOK, 128, 8, 1);
    k_gemm_naive<<<(NTOK * 128 + 255) / 256, 256, 0, stream>>>(
        gh, P[IN_W2_W], P[IN_W2_B], nullptr, 1.f, nullptr, gate, NTOK, 8, 128, 4);
    k_ifft2d_post<<<NP, 256, 0, stream>>>(fRe, fIm, gate, P[IN_NBN_G], P[IN_NBN_B], dst, 128);
  };

  int g128 = (NTOK * 128 + 255) / 256;
  int g64  = (NTOK * 64 + 255) / 256;

  // x0 = relu(bn(conv1(x)))
  gemm(x, P[IN_CONV1_W], nullptr, P[IN_CONV1_B], P[IN_CONV1_BN_G], BNK, P[IN_CONV1_BN_B],
       nullptr, nullptr, x0, 128, 128, 128, 0, 1);
  // x1 = x + cpe1(x)   (shortcut)
  k_dwconv3<<<g128, 256, 0, stream>>>(x, P[IN_CPE1_W], P[IN_CPE1_B], x1, 128, 1, 0);
  // fmt = FMT(x1)
  fmt_pipe(x1, fmt);
  // xs = LN1(x1)
  k_layernorm<<<NTOK / 8, 256, 0, stream>>>(x1, P[IN_LN1_G], P[IN_LN1_B], bA, NTOK);
  // dw3 branch
  gemm(bA, P[IN_DW3_PW_W], nullptr, P[IN_DW3_PW_B], nullptr, 1.f, nullptr, nullptr, nullptr,
       bB, 128, 128, 128, 0, 0);
  k_groupconv<<<g64, 256, 0, stream>>>(bB, P[IN_DW3_W], hA, 3);
  // dw5 branch
  gemm(bA, P[IN_DW5_PW_W], nullptr, P[IN_DW5_PW_B], nullptr, 1.f, nullptr, nullptr, nullptr,
       bB, 128, 128, 128, 0, 0);
  k_groupconv<<<g64, 256, 0, stream>>>(bB, P[IN_DW5_W], hB, 5);
  // act_res = silu(actproj(xs))
  gemm(bA, P[IN_ACTPROJ_W], nullptr, P[IN_ACTPROJ_B], nullptr, 1.f, nullptr, nullptr, nullptr,
       bC, 128, 128, 128, 0, 2);
  // xs3 = silu(dwc2(inproj2(xs3)))
  gemm(hA, P[IN_INPROJ2_W], nullptr, P[IN_INPROJ2_B], nullptr, 1.f, nullptr, nullptr, nullptr,
       hC, 64, 64, 64, 0, 0);
  k_dwconv3<<<g64, 256, 0, stream>>>(hC, P[IN_DWC2_W], P[IN_DWC2_B], hA, 64, 0, 2);
  // xs5 = silu(dwc2(inproj2(xs5)))
  gemm(hB, P[IN_INPROJ2_W], nullptr, P[IN_INPROJ2_B], nullptr, 1.f, nullptr, nullptr, nullptr,
       hC, 64, 64, 64, 0, 0);
  k_dwconv3<<<g64, 256, 0, stream>>>(hC, P[IN_DWC2_W], P[IN_DWC2_B], hB, 64, 0, 2);
  // attention(xs3): q=elu1(qW x), kv with elu1 on k-half
  gemm(hA, P[IN_Q_W], nullptr, P[IN_Q_B], nullptr, 1.f, nullptr, nullptr, nullptr, hC,
       64, 64, 64, 0, 5);
  gemm(hA, P[IN_KV_W], nullptr, P[IN_KV_B], nullptr, 1.f, nullptr, nullptr, nullptr, bB,
       64, 128, 64, 0, 6);
  k_attn_reduce<<<BDIM * 8, 256, 0, stream>>>(bB, ksum3, ctx3, 4096);
  k_attn_apply<<<(NTOK * 8 + 255) / 256, 256, 0, stream>>>(hC, ksum3, ctx3, hD, 4096);
  gemm(hD, P[IN_PR_W], nullptr, P[IN_PR_B], nullptr, 1.f, nullptr, nullptr, nullptr, hA,
       64, 64, 64, 0, 0);
  // attention(xs5)
  gemm(hB, P[IN_Q_W], nullptr, P[IN_Q_B], nullptr, 1.f, nullptr, nullptr, nullptr, hC,
       64, 64, 64, 0, 5);
  gemm(hB, P[IN_KV_W], nullptr, P[IN_KV_B], nullptr, 1.f, nullptr, nullptr, nullptr, bB,
       64, 128, 64, 0, 6);
  k_attn_reduce<<<BDIM * 8, 256, 0, stream>>>(bB, ksum5, ctx5, 4096);
  k_attn_apply<<<(NTOK * 8 + 255) / 256, 256, 0, stream>>>(hC, ksum5, ctx5, hD, 4096);
  gemm(hD, P[IN_PR_W], nullptr, P[IN_PR_B], nullptr, 1.f, nullptr, nullptr, nullptr, hB,
       64, 64, 64, 0, 0);
  // xs = concat(xs3,xs5) * act_res;  x2 = shortcut + outproj(xs) + fmt
  k_concat_mul<<<g128, 256, 0, stream>>>(hA, hB, bC, bB);
  gemm(bB, P[IN_OUTPROJ_W], nullptr, P[IN_OUTPROJ_B], nullptr, 1.f, nullptr, x1, fmt, bA,
       128, 128, 128, 0, 0);
  // x2 = x2 + cpe2(x2)
  k_dwconv3<<<g128, 256, 0, stream>>>(bA, P[IN_CPE2_W], P[IN_CPE2_B], bB, 128, 1, 0);
  // fmt2 = FMT(x2)
  fmt_pipe(bB, fmt);
  // ffn: h1 = gelu(LN2(x2) @ W1^T); x3 = x2 + h1 @ W2^T + fmt2
  k_layernorm<<<NTOK / 8, 256, 0, stream>>>(bB, P[IN_LN2_G], P[IN_LN2_B], bA, NTOK);
  gemm(bA, P[IN_FFN_W1], nullptr, nullptr, nullptr, 1.f, nullptr, nullptr, nullptr, h1b,
       128, 512, 128, 0, 3);
  gemm(h1b, P[IN_FFN_W2], nullptr, nullptr, nullptr, 1.f, nullptr, bB, fmt, bC,
       512, 128, 512, 0, 0);
  // xo = projout(x3)
  gemm(bC, P[IN_PROJOUT_W], nullptr, nullptr, nullptr, 1.f, nullptr, nullptr, nullptr, bA,
       128, 128, 128, 0, 0);
  // xr = relu(bn(reduce(concat(x0,xo)))) + x0  (split over weight halves)
  gemm(x0, P[IN_REDUCE_W], nullptr, nullptr, nullptr, 1.f, nullptr, nullptr, nullptr, bB,
       128, 128, 256, 0, 0);
  gemm(bA, P[IN_REDUCE_W], bB, P[IN_REDUCE_B], P[IN_REDUCE_BN_G], BNK, P[IN_REDUCE_BN_B],
       x0, nullptr, outp, 128, 128, 256, 128, 1);
}
